// MultiHeadSelfAttention_57432302682633
// MI455X (gfx1250) — matrix operand order repaired and checked by reading, not compiled, not run on hardware
//
#include <hip/hip_runtime.h>
#include <hip/hip_bf16.h>

#define EMBED 512
#define HEADS 8
#define HDIM  64
#define BATCH 2
#define SEQ   2048
#define NROWS (BATCH * SEQ)   // 4096

typedef __attribute__((ext_vector_type(16))) __bf16 v16bf;
typedef __attribute__((ext_vector_type(8)))  float  v8f;
typedef __attribute__((ext_vector_type(4)))  unsigned int u32x4;

struct Frag32B { u32x4 lo, hi; };

static __device__ __forceinline__ v16bf make_frag(u32x4 lo, u32x4 hi) {
    Frag32B t{lo, hi};
    return __builtin_bit_cast(v16bf, t);
}

static __device__ __forceinline__ unsigned short f32_to_bf16(float f) {
    unsigned int u = __builtin_bit_cast(unsigned int, f);
    u += 0x7FFFu + ((u >> 16) & 1u);   // round-to-nearest-even
    return (unsigned short)(u >> 16);
}

// A fragment (16x32 bf16) from row-major [M][K], tile origin (m0, k0).
// Lane L: row = m0 + L%16 ; K = k0 + 8*(L/16) + {0..7} and +16..+23.
static __device__ __forceinline__ v16bf load_a_frag(const unsigned short* __restrict__ p0,
                                                    int ld, int m0, int k0, int lane) {
    const unsigned short* p = p0 + (m0 + (lane & 15)) * ld + k0 + ((lane >> 4) << 3);
    u32x4 lo = *(const u32x4*)p;          // K: base .. base+7
    u32x4 hi = *(const u32x4*)(p + 16);   // K: base+16 .. base+23
    return make_frag(lo, hi);
}

// B fragment (32x16 bf16) from B^T storage [N][K] row-major, tile origin (k0, n0).
// Lane L: col N = n0 + L%16 ; K = k0 + 8*(L/16) + {0..7} and +16..+23.
static __device__ __forceinline__ v16bf load_b_frag(const unsigned short* __restrict__ p0,
                                                    int ld, int n0, int k0, int lane) {
    const unsigned short* p = p0 + (n0 + (lane & 15)) * ld + k0 + ((lane >> 4) << 3);
    u32x4 lo = *(const u32x4*)p;
    u32x4 hi = *(const u32x4*)(p + 16);
    return make_frag(lo, hi);
}

static __device__ __forceinline__ v8f wmma_bf16(v16bf a, v16bf b, v8f c) {
    return __builtin_amdgcn_wmma_f32_16x16x32_bf16(false, a, false, b, (short)0, c, false, false);
}

// ---------------------------------------------------------------- stage 1
__global__ void cvt_f32_bf16_kernel(const float* __restrict__ src,
                                    unsigned short* __restrict__ dst, int n) {
    int i = blockIdx.x * blockDim.x + threadIdx.x;
    if (i < n) dst[i] = f32_to_bf16(src[i]);
}

// Wt[n][k] = W[k][n]  (so GEMM B-fragments load contiguously)
__global__ void transpose_w_kernel(const float* __restrict__ W,
                                   unsigned short* __restrict__ Wt) {
    int i = blockIdx.x * 256 + threadIdx.x;   // over 512*512
    int n = i >> 9, k = i & 511;
    Wt[n * EMBED + k] = f32_to_bf16(W[k * EMBED + n]);
}

// ---------------------------------------------------------------- stage 2
// Fused Q/K/V projection, M-blocked by 4: one wave -> 64 rows x 16 cols of
// each of Q,K,V. B fragments loaded once per k-step, reused 4x (and A reused 3x).
// Q,K written as [b,h,s,d]; V written transposed [b,h,d,s].
__global__ __launch_bounds__(128) void qkv_kernel(
    const unsigned short* __restrict__ Xb,
    const unsigned short* __restrict__ Wqt,
    const unsigned short* __restrict__ Wkt,
    const unsigned short* __restrict__ Wvt,
    const float* __restrict__ bq, const float* __restrict__ bk,
    const float* __restrict__ bv,
    unsigned short* __restrict__ Q, unsigned short* __restrict__ K,
    unsigned short* __restrict__ Vt) {
    int lane = threadIdx.x & 31;
    int w  = blockIdx.x * 4 + (threadIdx.x >> 5);   // 2048 waves
    int mg = w >> 5;                                 // 64 row-groups of 64
    int nt = w & 31;
    int m0 = mg << 6, n0 = nt << 4;

    v8f cq[4], ck[4], cv[4];
    v8f zero = {};
#pragma unroll
    for (int mb = 0; mb < 4; ++mb) { cq[mb] = zero; ck[mb] = zero; cv[mb] = zero; }

    for (int k0 = 0; k0 < EMBED; k0 += 32) {
        v16bf bqf = load_b_frag(Wqt, EMBED, n0, k0, lane);
        v16bf bkf = load_b_frag(Wkt, EMBED, n0, k0, lane);
        v16bf bvf = load_b_frag(Wvt, EMBED, n0, k0, lane);
#pragma unroll
        for (int mb = 0; mb < 4; ++mb) {
            v16bf a = load_a_frag(Xb, EMBED, m0 + mb * 16, k0, lane);
            cq[mb] = wmma_bf16(a, bqf, cq[mb]);
            ck[mb] = wmma_bf16(a, bkf, ck[mb]);
            cv[mb] = wmma_bf16(a, bvf, cv[mb]);
        }
    }

    int n = n0 + (lane & 15);
    int h = n >> 6, d = n & 63;
    float bqv = bq[n], bkv = bk[n], bvv = bv[n];
#pragma unroll
    for (int mb = 0; mb < 4; ++mb) {
        int mbase = m0 + mb * 16 + ((lane >> 4) << 3);
#pragma unroll
        for (int r = 0; r < 8; ++r) {
            int s  = mbase + r;                  // global row 0..4095
            int b  = s >> 11, sl = s & (SEQ - 1);
            int bh = b * HEADS + h;
            Q[(bh * SEQ + sl) * HDIM + d]  = f32_to_bf16(cq[mb][r] + bqv);
            K[(bh * SEQ + sl) * HDIM + d]  = f32_to_bf16(ck[mb][r] + bkv);
            Vt[(bh * HDIM + d) * SEQ + sl] = f32_to_bf16(cv[mb][r] + bvv);
        }
    }
}

// ---------------------------------------------------------------- stage 3
// Flash attention: block = 4 waves x 16 query rows of one (b,h).
// K/V 64-key chunks staged once per workgroup into double-buffered LDS
// (global loads for chunk i+1 overlap WMMAs of chunk i).
__global__ __launch_bounds__(128) void attn_kernel(
    const unsigned short* __restrict__ Q, const unsigned short* __restrict__ K,
    const unsigned short* __restrict__ Vt, unsigned short* __restrict__ O) {
    __shared__ unsigned short Kbuf[2][64 * 64];   // [key_local][d]
    __shared__ unsigned short Vbuf[2][64 * 64];   // [d][key_local]
    __shared__ unsigned short Plds[4][16 * 64];

    int tid  = threadIdx.x;
    int lane = tid & 31;
    int wave = tid >> 5;
    int bh = blockIdx.x >> 5;                // 0..15
    int qt = blockIdx.x & 31;
    int q0 = (qt << 6) + (wave << 4);        // query row within head

    const unsigned short* Qh = Q  + (size_t)bh * SEQ * HDIM;
    const unsigned short* Kh = K  + (size_t)bh * SEQ * HDIM;
    const unsigned short* Vh = Vt + (size_t)bh * HDIM * SEQ;

    v16bf aq0 = load_a_frag(Qh, HDIM, q0, 0,  lane);
    v16bf aq1 = load_a_frag(Qh, HDIM, q0, 32, lane);

    float mrow[8], lrow[8];
    v8f o[4];
    v8f zero = {};
#pragma unroll
    for (int r = 0; r < 8; ++r) { mrow[r] = -3.0e38f; lrow[r] = 0.0f; }
#pragma unroll
    for (int t = 0; t < 4; ++t) o[t] = zero;

    unsigned short* Pw = &Plds[wave][0];
    const float scale = 0.125f;              // 1/sqrt(64)
    const int mloc = (lane >> 4) << 3;
    const int col  = lane & 15;

    u32x4 kst[4], vst[4];
    // ---- load + store chunk 0 (K chunk is 8 KB contiguous; V chunk is 64
    //      rows of 128 B at 4 KB stride)
#pragma unroll
    for (int j = 0; j < 4; ++j) {
        int u = j * 128 + tid;
        kst[j] = *(const u32x4*)(Kh + (size_t)u * 8);
        int d = u >> 3, cb = u & 7;
        vst[j] = *(const u32x4*)(Vh + (size_t)d * SEQ + cb * 8);
    }
#pragma unroll
    for (int j = 0; j < 4; ++j) {
        int u = j * 128 + tid;
        *(u32x4*)(&Kbuf[0][(size_t)u * 8]) = kst[j];
        int d = u >> 3, cb = u & 7;
        *(u32x4*)(&Vbuf[0][(size_t)d * 64 + cb * 8]) = vst[j];
    }

    for (int i = 0; i < SEQ / 64; ++i) {
        __syncthreads();                      // chunk i visible; buf i^1 free
        int cur = i & 1;
        int kc = i * 64;
        const unsigned short* Kc = &Kbuf[cur][0];
        const unsigned short* Vc = &Vbuf[cur][0];

        // prefetch chunk i+1 into registers (overlaps WMMAs below)
        if (i + 1 < SEQ / 64) {
            int kn = kc + 64;
#pragma unroll
            for (int j = 0; j < 4; ++j) {
                int u = j * 128 + tid;
                kst[j] = *(const u32x4*)(Kh + (size_t)kn * HDIM + (size_t)u * 8);
                int d = u >> 3, cb = u & 7;
                vst[j] = *(const u32x4*)(Vh + (size_t)d * SEQ + kn + cb * 8);
            }
        }

        // ---- logits tile: 16 q-rows x 64 keys (from LDS)
        v8f c[4];
#pragma unroll
        for (int t = 0; t < 4; ++t) {
            v8f acc = zero;
            acc = wmma_bf16(aq0, load_b_frag(Kc, HDIM, t * 16, 0,  lane), acc);
            acc = wmma_bf16(aq1, load_b_frag(Kc, HDIM, t * 16, 32, lane), acc);
            c[t] = acc * scale;
        }
        // ---- row max across tiles, then 16-lane butterfly within each half
        float mx[8];
#pragma unroll
        for (int r = 0; r < 8; ++r)
            mx[r] = fmaxf(fmaxf(c[0][r], c[1][r]), fmaxf(c[2][r], c[3][r]));
        for (int off = 1; off < 16; off <<= 1)
#pragma unroll
            for (int r = 0; r < 8; ++r)
                mx[r] = fmaxf(mx[r], __shfl_xor(mx[r], off, 32));

        float alpha[8], sum[8];
#pragma unroll
        for (int r = 0; r < 8; ++r) {
            float mn = fmaxf(mrow[r], mx[r]);
            alpha[r] = __expf(mrow[r] - mn);
            mrow[r]  = mn;
            sum[r]   = 0.0f;
        }
        // ---- P = exp(c - m), row sums, stash bf16 P into LDS (C -> A relayout)
#pragma unroll
        for (int t = 0; t < 4; ++t)
#pragma unroll
            for (int r = 0; r < 8; ++r) {
                float p = __expf(c[t][r] - mrow[r]);
                sum[r] += p;
                Pw[(mloc + r) * 64 + t * 16 + col] = f32_to_bf16(p);
            }
        for (int off = 1; off < 16; off <<= 1)
#pragma unroll
            for (int r = 0; r < 8; ++r)
                sum[r] += __shfl_xor(sum[r], off, 32);
#pragma unroll
        for (int r = 0; r < 8; ++r)
            lrow[r] = lrow[r] * alpha[r] + sum[r];
#pragma unroll
        for (int t = 0; t < 4; ++t)
#pragma unroll
            for (int r = 0; r < 8; ++r)
                o[t][r] *= alpha[r];

        // Intra-wave DS ops are in-order; clobber pins compiler ordering.
        asm volatile("s_wait_dscnt 0" ::: "memory");

        // ---- O += P (16x64) * V (64x64)  (V from LDS)
#pragma unroll
        for (int step = 0; step < 2; ++step) {
            v16bf ap = load_a_frag(Pw, 64, 0, step * 32, lane);
#pragma unroll
            for (int t = 0; t < 4; ++t)
                o[t] = wmma_bf16(ap,
                                 load_b_frag(Vc, HDIM, t * 16, step * 32, lane),
                                 o[t]);
        }

        // stage chunk i+1 into the other buffer (read again only after the
        // next barrier, and buf i^1 was last read before this iteration's
        // barrier -> safe without an extra sync)
        if (i + 1 < SEQ / 64) {
            int nxt = cur ^ 1;
#pragma unroll
            for (int j = 0; j < 4; ++j) {
                int u = j * 128 + tid;
                *(u32x4*)(&Kbuf[nxt][(size_t)u * 8]) = kst[j];
                int d = u >> 3, cb = u & 7;
                *(u32x4*)(&Vbuf[nxt][(size_t)d * 64 + cb * 8]) = vst[j];
            }
        }
    }

    // ---- epilogue: normalize and write merged-head [s][e] bf16
    int b = bh >> 3, h = bh & 7;
#pragma unroll
    for (int t = 0; t < 4; ++t) {
#pragma unroll
        for (int r = 0; r < 8; ++r) {
            float val = o[t][r] / lrow[r];
            int gs = b * SEQ + q0 + mloc + r;       // global row
            int e  = h * HDIM + t * 16 + col;
            O[(size_t)gs * EMBED + e] = f32_to_bf16(val);
        }
    }
}

// ---------------------------------------------------------------- stage 4
// Output projection, M-blocked by 4 (wave -> 64 rows x 16 cols), f32 output.
__global__ __launch_bounds__(128) void out_proj_kernel(
    const unsigned short* __restrict__ Ob, const unsigned short* __restrict__ Wot,
    const float* __restrict__ bo, float* __restrict__ out) {
    int lane = threadIdx.x & 31;
    int w  = blockIdx.x * 4 + (threadIdx.x >> 5);   // 2048 waves
    int mg = w >> 5, nt = w & 31;
    int m0 = mg << 6, n0 = nt << 4;

    v8f c[4];
    v8f zero = {};
#pragma unroll
    for (int mb = 0; mb < 4; ++mb) c[mb] = zero;

    for (int k0 = 0; k0 < EMBED; k0 += 32) {
        v16bf bf = load_b_frag(Wot, EMBED, n0, k0, lane);
#pragma unroll
        for (int mb = 0; mb < 4; ++mb) {
            v16bf a = load_a_frag(Ob, EMBED, m0 + mb * 16, k0, lane);
            c[mb] = wmma_bf16(a, bf, c[mb]);
        }
    }
    int n = n0 + (lane & 15);
    float bias = bo[n];
#pragma unroll
    for (int mb = 0; mb < 4; ++mb) {
        int mbase = m0 + mb * 16 + ((lane >> 4) << 3);
#pragma unroll
        for (int r = 0; r < 8; ++r)
            out[(size_t)(mbase + r) * EMBED + n] = c[mb][r] + bias;
    }
}

// ---------------------------------------------------------------- launch
extern "C" void kernel_launch(void* const* d_in, const int* in_sizes, int n_in,
                              void* d_out, int out_size, void* d_ws, size_t ws_size,
                              hipStream_t stream) {
    (void)in_sizes; (void)n_in; (void)out_size; (void)ws_size;

    const float* x  = (const float*)d_in[0];
    const float* Wq = (const float*)d_in[1];
    const float* bq = (const float*)d_in[2];
    const float* Wk = (const float*)d_in[3];
    const float* bk = (const float*)d_in[4];
    const float* Wv = (const float*)d_in[5];
    const float* bv = (const float*)d_in[6];
    const float* Wo = (const float*)d_in[7];
    const float* bo = (const float*)d_in[8];

    unsigned short* ws16 = (unsigned short*)d_ws;
    const size_t NX = (size_t)NROWS * EMBED;       // 2,097,152
    const size_t NW = (size_t)EMBED * EMBED;       //   262,144
    unsigned short* Xb  = ws16;                    // bf16 X
    unsigned short* Wqt = Xb  + NX;
    unsigned short* Wkt = Wqt + NW;
    unsigned short* Wvt = Wkt + NW;
    unsigned short* Wot = Wvt + NW;
    unsigned short* Qb  = Wot + NW;                // [b,h,s,d]
    unsigned short* Kb  = Qb  + NX;                // [b,h,s,d]
    unsigned short* Vtb = Kb  + NX;                // [b,h,d,s]
    unsigned short* Ob  = Vtb + NX;                // [s, e] merged heads

    // Stage 1: formats
    cvt_f32_bf16_kernel<<<(int)(NX / 256), 256, 0, stream>>>(x, Xb, (int)NX);
    transpose_w_kernel<<<(int)(NW / 256), 256, 0, stream>>>(Wq, Wqt);
    transpose_w_kernel<<<(int)(NW / 256), 256, 0, stream>>>(Wk, Wkt);
    transpose_w_kernel<<<(int)(NW / 256), 256, 0, stream>>>(Wv, Wvt);
    transpose_w_kernel<<<(int)(NW / 256), 256, 0, stream>>>(Wo, Wot);

    // Stage 2: fused QKV projection (64 row-groups * 32 n-tiles = 2048 waves)
    qkv_kernel<<<512, 128, 0, stream>>>(Xb, Wqt, Wkt, Wvt, bq, bk, bv, Qb, Kb, Vtb);

    // Stage 3: flash attention (16 heads * 32 q-tiles of 64 rows)
    attn_kernel<<<512, 128, 0, stream>>>(Qb, Kb, Vtb, Ob);

    // Stage 4: output projection (2048 waves)
    out_proj_kernel<<<512, 128, 0, stream>>>(Ob, Wot, bo, (float*)d_out);
}
